// WindowSA_644245094977
// MI455X (gfx1250) — compile-verified, matrix operand order checked by reading, not run on hardware
//
#include <hip/hip_runtime.h>
#include <hip/hip_bf16.h>
#include <math.h>

// Problem constants (fixed by harness setup_inputs)
#define NP_ROWS  524288      // N*P = 8192*64
#define C_DIM    128
#define M_WIN    4096
#define P_WIN    64
#define MW_ALL   262144      // M*P
#define A_ASY    131072
#define BL_BLK   65536

typedef __attribute__((ext_vector_type(16))) _Float16 v16h;
typedef __attribute__((ext_vector_type(8)))  float    v8f;

#define WMMA16(a, b, c) \
  __builtin_amdgcn_wmma_f32_16x16x32_f16(false, (a), false, (b), (short)0, (c), false, false)

__device__ __forceinline__ float wred_sum(float v) {
#pragma unroll
  for (int off = 16; off >= 1; off >>= 1) v += __shfl_xor(v, off, 32);
  return v;
}
__device__ __forceinline__ float wred_max(float v) {
#pragma unroll
  for (int off = 16; off >= 1; off >>= 1) v = fmaxf(v, __shfl_xor(v, off, 32));
  return v;
}

// A/B fragment from a row-major [rows][ld] f16 array.
// 16-bit A 16x32 layout: lanes 0-15 hold row r, K {k0+0..7, k0+16..23};
// lanes 16-31 hold row r, K {k0+8..15, k0+24..31}.  Two 16B loads per lane.
__device__ __forceinline__ v16h frag_ld_rm(const _Float16* __restrict__ base,
                                           int row0, int ld, int k0, int lane) {
  const int r    = lane & 15;
  const int koff = k0 + ((lane & 16) >> 1);   // +8 for upper half-wave
  const _Float16* p = base + (size_t)(row0 + r) * ld + koff;
  v16h f;
  ((uint4*)&f)[0] = *(const uint4*)(p);
  ((uint4*)&f)[1] = *(const uint4*)(p + 16);
  return f;
}

// L2 warm-up for a weight panel (emits global_prefetch_b8 path).
__device__ __forceinline__ void prefetch_panel(const _Float16* p, int bytes, int tid) {
  for (int off = tid * 128; off < bytes; off += 256 * 128)
    __builtin_prefetch((const char*)p + off, 0, 0);
}

// ---------------------------------------------------------------- flags
__global__ void k_zero_flags(unsigned char* a, unsigned char* b) {
  int t = blockIdx.x * blockDim.x + threadIdx.x;
  if (t < MW_ALL) { a[t] = 0; b[t] = 0; }
}
__global__ void k_set_flags(const int* __restrict__ asy, const int* __restrict__ blk,
                            unsigned char* af, unsigned char* bf) {
  int t = blockIdx.x * blockDim.x + threadIdx.x;
  if (t < A_ASY)  af[asy[t]] = 1;
  if (t < BL_BLK) bf[blk[t]] = 1;
}

// ---------------------------------------------------------------- weight convert (f32 -> f16, transposed to [N][K])
__global__ void k_cvt_weights(const float* __restrict__ qkvw, const float* __restrict__ projw,
                              const float* __restrict__ w1,   const float* __restrict__ w2,
                              _Float16* qkvWT, _Float16* projWT, _Float16* w1T, _Float16* w2T) {
  int t = blockIdx.x * blockDim.x + threadIdx.x;
  const int NQ = 128 * 384, NPJ = 128 * 128, NW1 = 128 * 512, NW2 = 512 * 128;
  if (t < NQ) {
    int k = t / 384, n = t % 384;
    qkvWT[n * 128 + k] = (_Float16)qkvw[t];
  } else if (t < NQ + NPJ) {
    int u = t - NQ; int k = u / 128, n = u % 128;
    projWT[n * 128 + k] = (_Float16)projw[u];
  } else if (t < NQ + NPJ + NW1) {
    int u = t - NQ - NPJ; int k = u / 512, n = u % 512;
    w1T[n * 128 + k] = (_Float16)w1[u];
  } else if (t < NQ + NPJ + NW1 + NW2) {
    int u = t - NQ - NPJ - NW1; int k = u / 128, n = u % 128;
    w2T[n * 512 + k] = (_Float16)w2[u];
  }
}

// ---------------------------------------------------------------- LN over all rows of x -> out (one wave per 128-elem row)
__launch_bounds__(256)
__global__ void k_ln_all(const float* __restrict__ x, const float* __restrict__ w,
                         const float* __restrict__ b, float* __restrict__ out) {
  const int wid  = (blockIdx.x << 3) + (threadIdx.x >> 5);
  const int lane = threadIdx.x & 31;
  float4 xv = *(const float4*)(x + (size_t)wid * C_DIM + lane * 4);
  float mu = wred_sum(xv.x + xv.y + xv.z + xv.w) * (1.0f / 128.0f);
  float d0 = xv.x - mu, d1 = xv.y - mu, d2 = xv.z - mu, d3 = xv.w - mu;
  float var = wred_sum(d0 * d0 + d1 * d1 + d2 * d2 + d3 * d3) * (1.0f / 128.0f);
  float inv = rsqrtf(var + 1e-6f);
  float4 wv = *(const float4*)(w + lane * 4);
  float4 bv = *(const float4*)(b + lane * 4);
  float4 o;
  o.x = d0 * inv * wv.x + bv.x;
  o.y = d1 * inv * wv.y + bv.y;
  o.z = d2 * inv * wv.z + bv.z;
  o.w = d3 * inv * wv.w + bv.w;
  *(float4*)(out + (size_t)wid * C_DIM + lane * 4) = o;
}

// ---------------------------------------------------------------- fused window attention
// One block (256 thr = 8 waves) per window m: gather 64 rows (re-LN asy rows),
// qkv GEMM, per-head QK^T + mask + softmax + PV, proj GEMM, scatter to XX.
__launch_bounds__(256)
__global__ void k_attn(const float* __restrict__ Xln, const int* __restrict__ idx_win,
                       const int* __restrict__ idx_part,
                       const unsigned char* __restrict__ asyf, const unsigned char* __restrict__ blkf,
                       const float* __restrict__ ln1w, const float* __restrict__ ln1b,
                       const _Float16* __restrict__ qkvWT, const float* __restrict__ qkvb,
                       const _Float16* __restrict__ projWT, const float* __restrict__ projb,
                       float* __restrict__ XX) {
  __shared__ __attribute__((aligned(16))) _Float16 Xh[64 * 144];   // reused as S f32 [64][72]
  __shared__ __attribute__((aligned(16))) _Float16 QKV[64 * 392];  // per-row [H=4][q32|k32|v32]
  __shared__ __attribute__((aligned(16))) _Float16 Ph[64 * 72];    // softmax probs (one head)
  __shared__ __attribute__((aligned(16))) _Float16 Vt[32 * 72];    // transposed V (one head): [d][k]
  __shared__ __attribute__((aligned(16))) _Float16 Oh[64 * 136];   // attention output (all heads)
  __shared__ int   rowIdx[64];
  __shared__ float blkMask[64];
  float* Sf = (float*)Xh;

  const int m    = blockIdx.x;
  const int tid  = threadIdx.x;
  const int wid  = tid >> 5;
  const int lane = tid & 31;

  prefetch_panel(qkvWT, 128 * 384 * 2, tid);
  prefetch_panel(projWT, 128 * 128 * 2, tid);

  // ---- stage A: gather rows, optional re-LN, f16 convert
  for (int p = wid; p < 64; p += 8) {
    const int gi = idx_part[m * 64 + p];
    if (lane == 0) { rowIdx[p] = gi; blkMask[p] = blkf[gi] ? 1.0f : 0.0f; }
    const int wrow = gi >> 6, pp = gi & 63;
    const float* src = Xln + ((size_t)idx_win[wrow] * 64 + pp) * C_DIM;
    float4 xv = *(const float4*)(src + lane * 4);
    if (asyf[gi]) {
      float mu = wred_sum(xv.x + xv.y + xv.z + xv.w) * (1.0f / 128.0f);
      float d0 = xv.x - mu, d1 = xv.y - mu, d2 = xv.z - mu, d3 = xv.w - mu;
      float var = wred_sum(d0 * d0 + d1 * d1 + d2 * d2 + d3 * d3) * (1.0f / 128.0f);
      float inv = rsqrtf(var + 1e-6f);
      float4 wv = *(const float4*)(ln1w + lane * 4);
      float4 bv = *(const float4*)(ln1b + lane * 4);
      xv.x = d0 * inv * wv.x + bv.x;
      xv.y = d1 * inv * wv.y + bv.y;
      xv.z = d2 * inv * wv.z + bv.z;
      xv.w = d3 * inv * wv.w + bv.w;
    }
    _Float16* dst = Xh + p * 144 + lane * 4;
    dst[0] = (_Float16)xv.x; dst[1] = (_Float16)xv.y;
    dst[2] = (_Float16)xv.z; dst[3] = (_Float16)xv.w;
  }
  __syncthreads();

  // ---- stage B: qkv GEMM 64x128 @ 128x384  (96 tiles, 12/wave)
  for (int t = wid; t < 96; t += 8) {
    const int mt = t / 24, nt = t % 24;
    v8f acc = {0.f, 0.f, 0.f, 0.f, 0.f, 0.f, 0.f, 0.f};
#pragma unroll
    for (int kk = 0; kk < 4; ++kk) {
      v16h a = frag_ld_rm(Xh, mt * 16, 144, kk * 32, lane);
      v16h b = frag_ld_rm(qkvWT, nt * 16, 128, kk * 32, lane);
      acc = WMMA16(a, b, acc);
    }
    const int n = nt * 16 + (lane & 15);
    const int mbase = mt * 16 + ((lane >> 4) << 3);
    const float bias = qkvb[n];
#pragma unroll
    for (int r = 0; r < 8; ++r)
      QKV[(mbase + r) * 392 + n] = (_Float16)(acc[r] + bias);
  }
  __syncthreads();

  const float scale = 0.17677669529663687f;  // 1/sqrt(32)

  for (int h = 0; h < 4; ++h) {
    // scores: 16 tiles (qt x kt), 2 per wave; single K=32 WMMA each
    for (int t = wid * 2; t < wid * 2 + 2; ++t) {
      const int qt = t >> 2, kt = t & 3;
      v16h a = frag_ld_rm(QKV + h * 96, qt * 16, 392, 0, lane);       // q rows
      v16h b = frag_ld_rm(QKV + h * 96 + 32, kt * 16, 392, 0, lane);  // k rows (B^T)
      v8f acc = {0.f, 0.f, 0.f, 0.f, 0.f, 0.f, 0.f, 0.f};
      acc = WMMA16(a, b, acc);
      const int n = kt * 16 + (lane & 15);
      const int mbase = qt * 16 + ((lane >> 4) << 3);
      const float blocked = blkMask[n];
#pragma unroll
      for (int r = 0; r < 8; ++r)
        Sf[(mbase + r) * 72 + n] = (blocked > 0.5f) ? -10000.0f : acc[r] * scale;
    }
    // transpose this head's V (64x32) into Vt[d][k] with vector stores
    {
      const int d  = tid >> 3;          // 0..31
      const int k0 = (tid & 7) * 8;     // 0..56
      _Float16 tmp[8];
#pragma unroll
      for (int j = 0; j < 8; ++j)
        tmp[j] = QKV[(k0 + j) * 392 + h * 96 + 64 + d];
      *(uint4*)(Vt + d * 72 + k0) = *(const uint4*)tmp;
    }
    __syncthreads();
    // softmax rows (8/wave, 2 cols/lane)
    for (int q = wid; q < 64; q += 8) {
      float a0 = Sf[q * 72 + lane], a1 = Sf[q * 72 + 32 + lane];
      float mx = wred_max(fmaxf(a0, a1));
      float e0 = __expf(a0 - mx), e1 = __expf(a1 - mx);
      float inv = 1.0f / wred_sum(e0 + e1);
      Ph[q * 72 + lane]      = (_Float16)(e0 * inv);
      Ph[q * 72 + 32 + lane] = (_Float16)(e1 * inv);
    }
    __syncthreads();
    // O = P @ V : 8 tiles (qt 0..3, dt 0..1), 1/wave, K=64
    {
      const int qt = wid >> 1, dt = wid & 1;
      v8f acc = {0.f, 0.f, 0.f, 0.f, 0.f, 0.f, 0.f, 0.f};
#pragma unroll
      for (int kk = 0; kk < 2; ++kk) {
        v16h a = frag_ld_rm(Ph, qt * 16, 72, kk * 32, lane);
        v16h b = frag_ld_rm(Vt, dt * 16, 72, kk * 32, lane);   // Vt is [d][k] row-major
        acc = WMMA16(a, b, acc);
      }
      const int n = dt * 16 + (lane & 15);
      const int mbase = qt * 16 + ((lane >> 4) << 3);
#pragma unroll
      for (int r = 0; r < 8; ++r)
        Oh[(mbase + r) * 136 + h * 32 + n] = (_Float16)acc[r];
    }
    __syncthreads();
  }

  // ---- proj: 64x128 @ 128x128 (32 tiles, 4/wave), scatter rows to XX[idx_part]
  for (int t = wid; t < 32; t += 8) {
    const int mt = t >> 3, nt = t & 7;
    v8f acc = {0.f, 0.f, 0.f, 0.f, 0.f, 0.f, 0.f, 0.f};
#pragma unroll
    for (int kk = 0; kk < 4; ++kk) {
      v16h a = frag_ld_rm(Oh, mt * 16, 136, kk * 32, lane);
      v16h b = frag_ld_rm(projWT, nt * 16, 128, kk * 32, lane);
      acc = WMMA16(a, b, acc);
    }
    const int n = nt * 16 + (lane & 15);
    const int mbase = mt * 16 + ((lane >> 4) << 3);
    const float bias = projb[n];
#pragma unroll
    for (int r = 0; r < 8; ++r)
      XX[(size_t)rowIdx[mbase + r] * C_DIM + n] = acc[r] + bias;
  }
}

// ---------------------------------------------------------------- fused MLP over asy rows (64 rows/block)
__launch_bounds__(256)
__global__ void k_mlp(const float* __restrict__ Xln, const int* __restrict__ idx_win,
                      const int* __restrict__ asy_idx,
                      const float* __restrict__ ln1w, const float* __restrict__ ln1b,
                      const float* __restrict__ gamma1,
                      const float* __restrict__ ln2w, const float* __restrict__ ln2b,
                      const _Float16* __restrict__ w1T, const float* __restrict__ b1,
                      const _Float16* __restrict__ w2T, const float* __restrict__ b2,
                      const float* __restrict__ gamma2,
                      float* __restrict__ XX) {
  __shared__ __attribute__((aligned(16))) float    Y[64 * 132];    // shortcut rows (f32)
  __shared__ __attribute__((aligned(16))) _Float16 Y2[64 * 144];   // LN2 rows (f16)
  __shared__ __attribute__((aligned(16))) _Float16 Hh[64 * 520];   // gelu hidden (f16)
  __shared__ int rowIdx[64];

  const int tid  = threadIdx.x;
  const int wid  = tid >> 5;
  const int lane = tid & 31;

  prefetch_panel(w1T, 128 * 512 * 2, tid);
  prefetch_panel(w2T, 512 * 128 * 2, tid);

  // ---- stage A: shortcut y = LN1(Xln row) + gamma1 * attn_out; y2 = LN2(y)
  for (int r = wid; r < 64; r += 8) {
    const int gi = asy_idx[blockIdx.x * 64 + r];
    if (lane == 0) rowIdx[r] = gi;
    const float* src = Xln + ((size_t)idx_win[gi >> 6] * 64 + (gi & 63)) * C_DIM;
    float4 xv = *(const float4*)(src + lane * 4);
    float mu = wred_sum(xv.x + xv.y + xv.z + xv.w) * (1.0f / 128.0f);
    float d0 = xv.x - mu, d1 = xv.y - mu, d2 = xv.z - mu, d3 = xv.w - mu;
    float var = wred_sum(d0 * d0 + d1 * d1 + d2 * d2 + d3 * d3) * (1.0f / 128.0f);
    float inv = rsqrtf(var + 1e-6f);
    float4 wv = *(const float4*)(ln1w + lane * 4);
    float4 bv = *(const float4*)(ln1b + lane * 4);
    float xa0 = d0 * inv * wv.x + bv.x;
    float xa1 = d1 * inv * wv.y + bv.y;
    float xa2 = d2 * inv * wv.z + bv.z;
    float xa3 = d3 * inv * wv.w + bv.w;
    float4 av = *(const float4*)(XX + (size_t)gi * C_DIM + lane * 4);
    float4 gv = *(const float4*)(gamma1 + lane * 4);
    float y0 = xa0 + gv.x * av.x;
    float y1 = xa1 + gv.y * av.y;
    float y2e = xa2 + gv.z * av.z;
    float y3 = xa3 + gv.w * av.w;
    Y[r * 132 + lane * 4 + 0] = y0;
    Y[r * 132 + lane * 4 + 1] = y1;
    Y[r * 132 + lane * 4 + 2] = y2e;
    Y[r * 132 + lane * 4 + 3] = y3;
    // LN2
    float mu2 = wred_sum(y0 + y1 + y2e + y3) * (1.0f / 128.0f);
    float e0 = y0 - mu2, e1 = y1 - mu2, e2 = y2e - mu2, e3 = y3 - mu2;
    float var2 = wred_sum(e0 * e0 + e1 * e1 + e2 * e2 + e3 * e3) * (1.0f / 128.0f);
    float inv2 = rsqrtf(var2 + 1e-6f);
    float4 w2v = *(const float4*)(ln2w + lane * 4);
    float4 b2v = *(const float4*)(ln2b + lane * 4);
    _Float16* dst = Y2 + r * 144 + lane * 4;
    dst[0] = (_Float16)(e0 * inv2 * w2v.x + b2v.x);
    dst[1] = (_Float16)(e1 * inv2 * w2v.y + b2v.y);
    dst[2] = (_Float16)(e2 * inv2 * w2v.z + b2v.z);
    dst[3] = (_Float16)(e3 * inv2 * w2v.w + b2v.w);
  }
  __syncthreads();

  // ---- GEMM1: 64x128 @ 128x512, gelu  (128 tiles, 16/wave)
  for (int t = wid; t < 128; t += 8) {
    const int mt = t >> 5, nt = t & 31;
    v8f acc = {0.f, 0.f, 0.f, 0.f, 0.f, 0.f, 0.f, 0.f};
#pragma unroll
    for (int kk = 0; kk < 4; ++kk) {
      v16h a = frag_ld_rm(Y2, mt * 16, 144, kk * 32, lane);
      v16h b = frag_ld_rm(w1T, nt * 16, 128, kk * 32, lane);
      acc = WMMA16(a, b, acc);
    }
    const int n = nt * 16 + (lane & 15);
    const int mbase = mt * 16 + ((lane >> 4) << 3);
    const float bias = b1[n];
#pragma unroll
    for (int r = 0; r < 8; ++r) {
      float v = acc[r] + bias;
      float g = 0.5f * v * (1.0f + erff(v * 0.70710678118654752f));  // exact gelu
      Hh[(mbase + r) * 520 + n] = (_Float16)g;
    }
  }
  __syncthreads();

  // ---- GEMM2: 64x512 @ 512x128 (+b2)*gamma2 + shortcut -> XX  (32 tiles, 4/wave)
  for (int t = wid; t < 32; t += 8) {
    const int mt = t >> 3, nt = t & 7;
    v8f acc = {0.f, 0.f, 0.f, 0.f, 0.f, 0.f, 0.f, 0.f};
#pragma unroll
    for (int kk = 0; kk < 16; ++kk) {
      v16h a = frag_ld_rm(Hh, mt * 16, 520, kk * 32, lane);
      v16h b = frag_ld_rm(w2T, nt * 16, 512, kk * 32, lane);
      acc = WMMA16(a, b, acc);
    }
    const int n = nt * 16 + (lane & 15);
    const int mbase = mt * 16 + ((lane >> 4) << 3);
    const float bias = b2[n], g2 = gamma2[n];
#pragma unroll
    for (int r = 0; r < 8; ++r) {
      const int row = mbase + r;
      XX[(size_t)rowIdx[row] * C_DIM + n] = Y[row * 132 + n] + (acc[r] + bias) * g2;
    }
  }
}

// ---------------------------------------------------------------- final scatter (skip blocked rows: they stay = LN(x))
__launch_bounds__(256)
__global__ void k_scatter(const float* __restrict__ XX, const int* __restrict__ idx_win,
                          const unsigned char* __restrict__ blkf, float* __restrict__ out) {
  const int i    = (blockIdx.x << 3) + (threadIdx.x >> 5);
  const int lane = threadIdx.x & 31;
  if (blkf[i]) return;
  float4 v = *(const float4*)(XX + (size_t)i * C_DIM + lane * 4);
  const int w = i >> 6, p = i & 63;
  *(float4*)(out + ((size_t)idx_win[w] * 64 + p) * C_DIM + lane * 4) = v;
}

// ---------------------------------------------------------------- launch
extern "C" void kernel_launch(void* const* d_in, const int* in_sizes, int n_in,
                              void* d_out, int out_size, void* d_ws, size_t ws_size,
                              hipStream_t stream) {
  const float* x        = (const float*)d_in[0];
  const int*   idx_win  = (const int*)d_in[1];
  const int*   idx_part = (const int*)d_in[2];
  const int*   asy      = (const int*)d_in[3];
  const int*   blk      = (const int*)d_in[4];
  const float* ln1w     = (const float*)d_in[7];
  const float* ln1b     = (const float*)d_in[8];
  const float* qkvw     = (const float*)d_in[9];
  const float* qkvb     = (const float*)d_in[10];
  const float* projw    = (const float*)d_in[11];
  const float* projb    = (const float*)d_in[12];
  const float* gamma1   = (const float*)d_in[13];
  const float* ln2w     = (const float*)d_in[14];
  const float* ln2b     = (const float*)d_in[15];
  const float* w1       = (const float*)d_in[16];
  const float* b1       = (const float*)d_in[17];
  const float* w2       = (const float*)d_in[18];
  const float* b2       = (const float*)d_in[19];
  const float* gamma2   = (const float*)d_in[20];
  float* out = (float*)d_out;

  char* ws = (char*)d_ws;
  float*         XX     = (float*)ws;                                   // 128 MB
  unsigned char* asyf   = (unsigned char*)(ws + (size_t)134217728);     // 256 KB
  unsigned char* blkf   = asyf + MW_ALL;                                // 256 KB
  _Float16*      qkvWT  = (_Float16*)(ws + (size_t)134217728 + 524288); // [384][128]
  _Float16*      projWT = qkvWT + 128 * 384;                            // [128][128]
  _Float16*      w1T    = projWT + 128 * 128;                           // [512][128]
  _Float16*      w2T    = w1T + 128 * 512;                              // [128][512]

  k_zero_flags<<<(MW_ALL + 255) / 256, 256, 0, stream>>>(asyf, blkf);
  k_set_flags<<<(A_ASY + 255) / 256, 256, 0, stream>>>(asy, blk, asyf, blkf);
  k_cvt_weights<<<(196608 + 255) / 256, 256, 0, stream>>>(qkvw, projw, w1, w2,
                                                          qkvWT, projWT, w1T, w2T);
  k_ln_all<<<NP_ROWS / 8, 256, 0, stream>>>(x, ln1w, ln1b, out);
  k_attn<<<M_WIN, 256, 0, stream>>>(out, idx_win, idx_part, asyf, blkf,
                                    ln1w, ln1b, qkvWT, qkvb, projWT, projb, XX);
  k_mlp<<<A_ASY / 64, 256, 0, stream>>>(out, idx_win, asy, ln1w, ln1b, gamma1,
                                        ln2w, ln2b, w1T, b1, w2T, b2, gamma2, XX);
  k_scatter<<<MW_ALL / 8, 256, 0, stream>>>(XX, idx_win, blkf, out);
}
